// ABTEModule_41205916238120
// MI455X (gfx1250) — compile-verified, matrix operand order checked by reading, not run on hardware
//
#include <hip/hip_runtime.h>
#include <hip/hip_bf16.h>

// ---------------------------------------------------------------------------
// CDNA5 (gfx1250) fused implementation of the ABTE module.
// All GEMMs run on the matrix engine via V_WMMA_F32_16X16X32_BF16 (wave32).
// Key algebraic optimizations:
//  * tsv branch computed once per (bucket, t)  -> 800 rows instead of 400k
//  * sc_w1 [384,256] split into nf-part (rows 0..255) and tf-part (256..383)
//  * sc_b1 folded into the precomputed tf-part; sc_b2 cancels in softmax
// Perf structure:
//  * node kernel processes 32 nodes / WG: each packed B fragment loaded once
//    from L2 feeds two WMMAs (halves L2 weight traffic vs 16-node tiles)
// ---------------------------------------------------------------------------

typedef __attribute__((ext_vector_type(16))) __bf16 v16bf;
typedef __attribute__((ext_vector_type(8)))  __bf16 v8bf;
typedef __attribute__((ext_vector_type(8)))  float  v8f;

union FragU { v16bf v; struct { v8bf lo; v8bf hi; } s; };

__device__ __forceinline__ v8f wmma_bf16(v16bf a, v16bf b, v8f c) {
  // D = A(16x32) * B(32x16) + C(16x16), f32 accumulate
  return __builtin_amdgcn_wmma_f32_16x16x32_bf16(
      /*neg_a=*/false, a, /*neg_b=*/false, b,
      /*c_mod=*/(short)0, c, /*reuse_a=*/false, /*reuse_b=*/false);
}

// A fragment from LDS, row-major bf16 [16][strideHalves].
// ISA 16-bit A layout: lane m = lane%16, group g = lane/16,
// half h -> K = ktile*32 + g*8 + (h<8 ? h : h+8)  => two 16B contiguous reads.
__device__ __forceinline__ v16bf load_afrag(const __bf16* base, int strideHalves,
                                            int lane, int ktile) {
  int m = lane & 15, g = lane >> 4;
  const __bf16* p = base + m * strideHalves + ktile * 32 + g * 8;
  FragU u;
  u.s.lo = *(const v8bf*)(p);
  u.s.hi = *(const v8bf*)(p + 16);
  return u.v;
}

// B fragment pre-packed by pack_kernel: fragment = 32 lanes x 16 halves,
// lane-contiguous -> one 32-byte load per lane.
__device__ __forceinline__ v16bf load_bfrag(const __bf16* packed, int fragIdx, int lane) {
  const __bf16* p = packed + ((size_t)fragIdx * 32 + (size_t)lane) * 16;
  return *(const v16bf*)p;
}

// ---------------------------------------------------------------------------
// Pack fp32 weight W (rows rowOffset..rowOffset+K-1 of a [*,Nn] matrix) into
// bf16 WMMA B-fragments.  fragIdx = nt*(K/32)+kt ; element ((frag*32+lane)*16+h)
// holds W[kt*32 + g*8 + (h<8?h:h+8), nt*16 + lane%16].
// ---------------------------------------------------------------------------
__global__ void pack_kernel(const float* __restrict__ W, __bf16* __restrict__ out,
                            int K, int Nn, int rowOffset) {
  int total = K * Nn;
  int idx = blockIdx.x * blockDim.x + threadIdx.x;
  if (idx >= total) return;
  int h    = idx & 15;
  int lane = (idx >> 4) & 31;
  int frag = idx >> 9;
  int Kt = K >> 5;
  int nt = frag / Kt;
  int kt = frag - nt * Kt;
  int n  = nt * 16 + (lane & 15);
  int g  = lane >> 4;
  int kl = (h < 8 ? h : h + 8) + g * 8;
  int k  = kt * 32 + kl;
  out[idx] = (__bf16)W[(long)(k + rowOffset) * Nn + n];
}

// ---------------------------------------------------------------------------
// TSV branch: for each of the B*T unique (bucket,t) rows compute
//   x(256) -> Linear(128) -> ReLU -> LN -> Linear(128) -> (@ sc_w1[256:]) + sc_b1
// and store tfsc[B*T, 256] (f32) to workspace.
// 16 rows per workgroup, 8 wave32.
// ---------------------------------------------------------------------------
__global__ __launch_bounds__(256)
void tsv_kernel(const float* __restrict__ tsv,
                const float* __restrict__ tp_b1, const float* __restrict__ tp_g,
                const float* __restrict__ tp_be, const float* __restrict__ tp_b2,
                const float* __restrict__ sc_b1,
                const __bf16* __restrict__ w1p,   // tp_w1 packed   256x128
                const __bf16* __restrict__ w2p,   // tp_w2 packed   128x128
                const __bf16* __restrict__ w3p,   // sc_w1[256:]    128x256
                float* __restrict__ tfsc, int Bk, int T) {
  __shared__ __align__(16) __bf16 Abuf[16 * 264];  // x tile, bf16
  __shared__ __align__(16) float  Hbuf[16 * 136];  // relu(l1)
  __shared__ __align__(16) __bf16 Lbuf[16 * 136];  // LN result
  __shared__ __align__(16) __bf16 Tbuf[16 * 136];  // tf (l2 out)

  int tid = threadIdx.x, lane = tid & 31, wave = tid >> 5;
  int R = Bk * T;
  int row0 = blockIdx.x * 16;

  // stage X[16,256] as bf16
  for (int i = tid; i < 16 * 256; i += 256) {
    int m = i >> 8, c = i & 255;
    int r = row0 + m;
    float v = 0.f;
    if (r < R) { int b = r / T, t = r - b * T; v = tsv[((long)t * Bk + b) * 256 + c]; }
    Abuf[m * 264 + c] = (__bf16)v;
  }
  __syncthreads();

  // GEMM1: [16,256]x[256,128], 8 ktiles, 8 ntiles (one per wave)
  {
    int nt = wave;
    v8f acc = {};
    for (int kt = 0; kt < 8; ++kt)
      acc = wmma_bf16(load_afrag(Abuf, 264, lane, kt), load_bfrag(w1p, nt * 8 + kt, lane), acc);
    int n = nt * 16 + (lane & 15), g = lane >> 4;
    float bias = tp_b1[n];
    for (int r = 0; r < 8; ++r) {
      float v = acc[r] + bias;
      Hbuf[(r + g * 8) * 136 + n] = v > 0.f ? v : 0.f;
    }
  }
  __syncthreads();

  // LayerNorm over 128 (16 threads per row)
  {
    int m = tid >> 4, sub = tid & 15;
    float s = 0.f, s2 = 0.f;
    for (int j = 0; j < 8; ++j) { float v = Hbuf[m * 136 + sub + j * 16]; s += v; s2 += v * v; }
    for (int off = 1; off < 16; off <<= 1) { s += __shfl_xor(s, off, 16); s2 += __shfl_xor(s2, off, 16); }
    float mu = s * (1.f / 128.f);
    float inv = rsqrtf(s2 * (1.f / 128.f) - mu * mu + 1e-5f);
    for (int j = 0; j < 8; ++j) {
      int c = sub + j * 16;
      Lbuf[m * 136 + c] = (__bf16)((Hbuf[m * 136 + c] - mu) * inv * tp_g[c] + tp_be[c]);
    }
  }
  __syncthreads();

  // GEMM2: [16,128]x[128,128]
  {
    int nt = wave;
    v8f acc = {};
    for (int kt = 0; kt < 4; ++kt)
      acc = wmma_bf16(load_afrag(Lbuf, 136, lane, kt), load_bfrag(w2p, nt * 4 + kt, lane), acc);
    int n = nt * 16 + (lane & 15), g = lane >> 4;
    float bias = tp_b2[n];
    for (int r = 0; r < 8; ++r)
      Tbuf[(r + g * 8) * 136 + n] = (__bf16)(acc[r] + bias);
  }
  __syncthreads();

  // GEMM3: tf @ sc_w1[256:384]  -> [16,256], fold sc_b1
  for (int half = 0; half < 2; ++half) {
    int nt = wave + half * 8;
    v8f acc = {};
    for (int kt = 0; kt < 4; ++kt)
      acc = wmma_bf16(load_afrag(Tbuf, 136, lane, kt), load_bfrag(w3p, nt * 4 + kt, lane), acc);
    int n = nt * 16 + (lane & 15), g = lane >> 4;
    float bias = sc_b1[n];
    for (int r = 0; r < 8; ++r) {
      int rr = row0 + r + g * 8;
      if (rr < R) tfsc[(long)rr * 256 + n] = acc[r] + bias;
    }
  }
}

// ---------------------------------------------------------------------------
// Node branch + scorer + softmax + aggregation, fused. 32 nodes / workgroup:
// each packed B fragment is loaded once and used by two WMMAs (M-tiles).
// ---------------------------------------------------------------------------
__global__ __launch_bounds__(256)
void node_kernel(const float* __restrict__ emb, const float* __restrict__ stats,
                 const float* __restrict__ teacher, const int* __restrict__ buckets,
                 const float* __restrict__ np_b1, const float* __restrict__ np_g,
                 const float* __restrict__ np_be, const float* __restrict__ np_b2,
                 const float* __restrict__ sc_w2, const float* __restrict__ tfsc,
                 const __bf16* __restrict__ w1p,   // np_w1 packed   320x512
                 const __bf16* __restrict__ w2p,   // np_w2 packed   512x256
                 const __bf16* __restrict__ w3p,   // sc_w1[:256]    256x256
                 float* __restrict__ out_agg, float* __restrict__ out_w,
                 int N, int T) {
  __shared__ __align__(16) __bf16 Abuf[32 * 328];  // node features bf16
  __shared__ __align__(16) float  Hbuf[32 * 520];  // relu(l1)  f32
  __shared__ __align__(16) __bf16 Lbuf[32 * 520];  // LN result bf16
  __shared__ __align__(16) __bf16 NFbuf[32 * 264]; // nf        bf16
  __shared__ __align__(16) float  Sbuf[32 * 264];  // nf @ sc_w1_top
  __shared__ float scores[32 * 8];
  __shared__ float wbuf[32 * 8];
  __shared__ int   bkt[32];

  int tid = threadIdx.x, lane = tid & 31, wave = tid >> 5;
  int n0 = blockIdx.x * 32;
  int Tc = T > 8 ? 8 : T;

  // stage features: [emb(256) | stats(64)] for 32 nodes
  for (int i = tid; i < 32 * 320; i += 256) {
    int m = i / 320, c = i - m * 320;
    int n = n0 + m;
    float v = 0.f;
    if (n < N) v = (c < 256) ? emb[(long)n * 256 + c] : stats[(long)n * 64 + (c - 256)];
    Abuf[m * 328 + c] = (__bf16)v;
  }
  if (tid < 32) {
    int n = n0 + tid;
    bkt[tid] = (n < N) ? (buckets[n] - 1) : 0;
  }
  __syncthreads();

  // GEMM1: [32,320]x[320,512]; 10 ktiles; 32 ntiles -> 4 per wave; 2 M-tiles
  for (int q = 0; q < 4; ++q) {
    int nt = wave + q * 8;
    v8f acc0 = {}, acc1 = {};
    for (int kt = 0; kt < 10; ++kt) {
      v16bf b = load_bfrag(w1p, nt * 10 + kt, lane);
      acc0 = wmma_bf16(load_afrag(Abuf,            328, lane, kt), b, acc0);
      acc1 = wmma_bf16(load_afrag(Abuf + 16 * 328, 328, lane, kt), b, acc1);
    }
    int n = nt * 16 + (lane & 15), g = lane >> 4;
    float bias = np_b1[n];
    for (int r = 0; r < 8; ++r) {
      float v0 = acc0[r] + bias, v1 = acc1[r] + bias;
      Hbuf[(r + g * 8) * 520 + n]        = v0 > 0.f ? v0 : 0.f;
      Hbuf[(r + g * 8 + 16) * 520 + n]   = v1 > 0.f ? v1 : 0.f;
    }
  }
  __syncthreads();

  // LayerNorm over 512 (8 threads per row, 32 rows)
  {
    int m = tid >> 3, sub = tid & 7;
    float s = 0.f, s2 = 0.f;
    for (int j = 0; j < 64; ++j) { float v = Hbuf[m * 520 + sub + j * 8]; s += v; s2 += v * v; }
    for (int off = 1; off < 8; off <<= 1) { s += __shfl_xor(s, off, 8); s2 += __shfl_xor(s2, off, 8); }
    float mu = s * (1.f / 512.f);
    float inv = rsqrtf(s2 * (1.f / 512.f) - mu * mu + 1e-5f);
    for (int j = 0; j < 64; ++j) {
      int c = sub + j * 8;
      Lbuf[m * 520 + c] = (__bf16)((Hbuf[m * 520 + c] - mu) * inv * np_g[c] + np_be[c]);
    }
  }
  __syncthreads();

  // GEMM2: [32,512]x[512,256]; 16 ktiles; 16 ntiles -> 2 per wave; 2 M-tiles
  for (int half = 0; half < 2; ++half) {
    int nt = wave + half * 8;
    v8f acc0 = {}, acc1 = {};
    for (int kt = 0; kt < 16; ++kt) {
      v16bf b = load_bfrag(w2p, nt * 16 + kt, lane);
      acc0 = wmma_bf16(load_afrag(Lbuf,            520, lane, kt), b, acc0);
      acc1 = wmma_bf16(load_afrag(Lbuf + 16 * 520, 520, lane, kt), b, acc1);
    }
    int n = nt * 16 + (lane & 15), g = lane >> 4;
    float bias = np_b2[n];
    for (int r = 0; r < 8; ++r) {
      NFbuf[(r + g * 8) * 264 + n]      = (__bf16)(acc0[r] + bias);
      NFbuf[(r + g * 8 + 16) * 264 + n] = (__bf16)(acc1[r] + bias);
    }
  }
  __syncthreads();

  // GEMM3: nf @ sc_w1[:256] -> [32,256]; 8 ktiles; 16 ntiles -> 2 per wave
  for (int half = 0; half < 2; ++half) {
    int nt = wave + half * 8;
    v8f acc0 = {}, acc1 = {};
    for (int kt = 0; kt < 8; ++kt) {
      v16bf b = load_bfrag(w3p, nt * 8 + kt, lane);
      acc0 = wmma_bf16(load_afrag(NFbuf,            264, lane, kt), b, acc0);
      acc1 = wmma_bf16(load_afrag(NFbuf + 16 * 264, 264, lane, kt), b, acc1);
    }
    int n = nt * 16 + (lane & 15), g = lane >> 4;
    for (int r = 0; r < 8; ++r) {
      Sbuf[(r + g * 8) * 264 + n]      = acc0[r];
      Sbuf[(r + g * 8 + 16) * 264 + n] = acc1[r];
    }
  }
  __syncthreads();

  // scorer: s[m,t] = relu(Sbuf[m,:] + tfsc[bkt[m]*T+t,:]) . sc_w2
  // 32 nodes x 8 t = 256 pairs -> one per thread
  {
    int m = tid >> 3, t = tid & 7;
    if (t < Tc) {
      const float* tp = tfsc + ((long)bkt[m] * T + t) * 256;
      float s = 0.f;
      for (int c = 0; c < 256; ++c) {
        float v = Sbuf[m * 264 + c] + tp[c];
        if (v > 0.f) s += v * sc_w2[c];
      }
      scores[m * 8 + t] = s;
    }
  }
  __syncthreads();

  // softmax over T per node (sc_b2 is a constant shift -> cancels)
  if (tid < 32) {
    int n = n0 + tid;
    float mx = -1e30f;
    for (int t = 0; t < Tc; ++t) mx = fmaxf(mx, scores[tid * 8 + t]);
    float wv[8];
    float sum = 0.f;
    for (int t = 0; t < Tc; ++t) { wv[t] = __expf(scores[tid * 8 + t] - mx); sum += wv[t]; }
    float inv = 1.f / sum;
    for (int t = 0; t < Tc; ++t) {
      float wgt = wv[t] * inv;
      wbuf[tid * 8 + t] = wgt;
      if (n < N) out_w[(long)n * T + t] = wgt;
    }
  }
  __syncthreads();

  // agg[n,c] = sum_t w[n,t] * teacher[n,t,c]   (10 classes, 320 outputs)
  for (int i = tid; i < 32 * 10; i += 256) {
    int m = i / 10, c = i - m * 10;
    int n = n0 + m;
    if (n < N) {
      float a = 0.f;
      for (int t = 0; t < Tc; ++t)
        a += wbuf[m * 8 + t] * teacher[((long)n * T + t) * 10 + c];
      out_agg[(long)n * 10 + c] = a;
    }
  }
}

// ---------------------------------------------------------------------------
extern "C" void kernel_launch(void* const* d_in, const int* in_sizes, int n_in,
                              void* d_out, int out_size, void* d_ws, size_t ws_size,
                              hipStream_t stream) {
  const float* tsv        = (const float*)d_in[0];
  const float* node_emb   = (const float*)d_in[1];
  const float* node_stats = (const float*)d_in[2];
  const float* teacher    = (const float*)d_in[3];
  const int*   buckets    = (const int*)  d_in[4];
  const float* tp_w1 = (const float*)d_in[5];
  const float* tp_b1 = (const float*)d_in[6];
  const float* tp_g  = (const float*)d_in[7];
  const float* tp_be = (const float*)d_in[8];
  const float* tp_w2 = (const float*)d_in[9];
  const float* tp_b2 = (const float*)d_in[10];
  const float* np_w1 = (const float*)d_in[11];
  const float* np_b1 = (const float*)d_in[12];
  const float* np_g  = (const float*)d_in[13];
  const float* np_be = (const float*)d_in[14];
  const float* np_w2 = (const float*)d_in[15];
  const float* np_b2 = (const float*)d_in[16];
  const float* sc_w1 = (const float*)d_in[17];
  const float* sc_b1 = (const float*)d_in[18];
  const float* sc_w2 = (const float*)d_in[19];
  // sc_b2 (d_in[20]) cancels in the softmax -> unused.

  const int N  = in_sizes[1] / 256;
  const int T  = in_sizes[3] / (N * 10);
  const int Bk = in_sizes[0] / (T * 256);
  const int R  = Bk * T;

  // workspace layout (byte offsets, all 1KB aligned)
  char* ws = (char*)d_ws;
  __bf16* np_w1p     = (__bf16*)(ws + 0);        // 320x512 -> 327680 B
  __bf16* np_w2p     = (__bf16*)(ws + 327680);   // 512x256 -> 262144 B
  __bf16* sc_w1top_p = (__bf16*)(ws + 589824);   // 256x256 -> 131072 B
  __bf16* tp_w1p     = (__bf16*)(ws + 720896);   // 256x128 ->  65536 B
  __bf16* tp_w2p     = (__bf16*)(ws + 786432);   // 128x128 ->  32768 B
  __bf16* sc_w1bot_p = (__bf16*)(ws + 819200);   // 128x256 ->  65536 B
  float*  tfsc       = (float*)(ws + 884736);    // R*256*4 B

  float* out_agg = (float*)d_out;          // [N,10]
  float* out_w   = out_agg + (long)N * 10; // [N,T]

  auto packLaunch = [&](const float* W, __bf16* outp, int K, int Nn, int rowOffset) {
    int total = K * Nn;
    pack_kernel<<<(total + 255) / 256, 256, 0, stream>>>(W, outp, K, Nn, rowOffset);
  };
  packLaunch(np_w1, np_w1p,     320, 512, 0);
  packLaunch(np_w2, np_w2p,     512, 256, 0);
  packLaunch(sc_w1, sc_w1top_p, 256, 256, 0);
  packLaunch(tp_w1, tp_w1p,     256, 128, 0);
  packLaunch(tp_w2, tp_w2p,     128, 128, 0);
  packLaunch(sc_w1, sc_w1bot_p, 128, 256, 256);  // bottom 128 rows of sc_w1

  tsv_kernel<<<(R + 15) / 16, 256, 0, stream>>>(
      tsv, tp_b1, tp_g, tp_be, tp_b2, sc_b1,
      tp_w1p, tp_w2p, sc_w1bot_p, tfsc, Bk, T);

  node_kernel<<<(N + 31) / 32, 256, 0, stream>>>(
      node_emb, node_stats, teacher, buckets,
      np_b1, np_g, np_be, np_b2, sc_w2, tfsc,
      np_w1p, np_w2p, sc_w1top_p, out_agg, out_w, N, T);
}
